// MOE_45148696217036
// MI455X (gfx1250) — matrix operand order repaired and checked by reading, not compiled, not run on hardware
//
#include <hip/hip_runtime.h>
#include <stdint.h>

// ---------------------------------------------------------------------------
// CDNA5 (gfx1250) MoE forward: top-2 routed experts + shared expert.
//  - all GEMMs: v_wmma_f32_16x16x32_bf16, wave32, 128x256x32 block tiles
//  - global->LDS staging: GLOBAL_LOAD_ASYNC_TO_LDS_B128, double buffered,
//    with all gather addresses hoisted out of the k-loop
//  - weights pre-converted to bf16 and pre-transposed to [N][K] once per call
// ---------------------------------------------------------------------------

typedef __attribute__((ext_vector_type(16))) __bf16 v16bf;
typedef __attribute__((ext_vector_type(8)))  float  v8f;
typedef int v4i __attribute__((vector_size(16)));

#define TILE_M 128
#define TILE_N 256
#define TILE_K 32
#define LDSS   40                       // TILE_K + 8 pad (bf16) -> 80B stride
#define BUF_ELEMS ((TILE_M + TILE_N) * LDSS)

// ---- async global->LDS copy (16B per lane), ASYNCcnt-tracked ---------------
#if defined(__has_builtin)
#  if __has_builtin(__builtin_amdgcn_global_load_async_to_lds_b128)
#    define HAVE_ASYNC_BUILTIN 1
#  endif
#endif

#ifdef HAVE_ASYNC_BUILTIN
#define ASYNC_CP(ldsptr, gptr)                                              \
  __builtin_amdgcn_global_load_async_to_lds_b128(                           \
      (__attribute__((address_space(1))) v4i*)(gptr),                       \
      (__attribute__((address_space(3))) v4i*)(ldsptr), 0, 0)
#else
#define ASYNC_CP(ldsptr, gptr)                                              \
  asm volatile("global_load_async_to_lds_b128 %0, %1, off" ::               \
               "v"((unsigned)(unsigned long long)(ldsptr)),                 \
               "v"((unsigned long long)(gptr)) : "memory")
#endif

#if defined(__has_builtin) && __has_builtin(__builtin_amdgcn_s_wait_asynccnt)
#define WAIT_ASYNC0() __builtin_amdgcn_s_wait_asynccnt(0)
#else
#define WAIT_ASYNC0() asm volatile("s_wait_asynccnt 0x0" ::: "memory")
#endif

static __device__ __forceinline__ unsigned short f2bf(float f) {
  unsigned u = __float_as_uint(f);
  unsigned r = 0x7FFFu + ((u >> 16) & 1u);
  return (unsigned short)((u + r) >> 16);
}

union FragU { v16bf v; uint4 q[2]; };

// ---------------------------------------------------------------------------
// helper kernels
// ---------------------------------------------------------------------------
__global__ void moe_zero_counts(int* counts, int e) {
  if ((int)threadIdx.x < e) counts[threadIdx.x] = 0;
}

__global__ void moe_f32_to_bf16(const float* __restrict__ src,
                                unsigned short* __restrict__ dst, long n8) {
  long i = (long)blockIdx.x * 256 + threadIdx.x;
  if (i >= n8) return;
  const float4* s = (const float4*)src + i * 2;
  float4 a = s[0], b = s[1];
  uint4 o;
  o.x = (unsigned)f2bf(a.x) | ((unsigned)f2bf(a.y) << 16);
  o.y = (unsigned)f2bf(a.z) | ((unsigned)f2bf(a.w) << 16);
  o.z = (unsigned)f2bf(b.x) | ((unsigned)f2bf(b.y) << 16);
  o.w = (unsigned)f2bf(b.z) | ((unsigned)f2bf(b.w) << 16);
  ((uint4*)dst)[i] = o;
}

// convert + transpose: src fp32 [K][N] (row-major) -> dst bf16 [N][K]
// grid: (N/32, K/32, batch), block (32, 8)
__global__ void moe_transpose_bf16(const float* __restrict__ src,
                                   unsigned short* __restrict__ dst,
                                   int K, int N) {
  __shared__ unsigned short t[32][33];
  const long bo = (long)blockIdx.z * K * N;
  const float* s = src + bo;
  unsigned short* d = dst + bo;
  const int n0 = blockIdx.x * 32, k0 = blockIdx.y * 32;
  const int tx = threadIdx.x, ty = threadIdx.y;
#pragma unroll
  for (int i = 0; i < 4; ++i) {
    int k = ty + i * 8;
    t[tx][k] = f2bf(s[(long)(k0 + k) * N + n0 + tx]);
  }
  __syncthreads();
#pragma unroll
  for (int i = 0; i < 4; ++i) {
    int n = ty + i * 8;
    d[(long)(n0 + n) * K + k0 + tx] = t[n][tx];
  }
}

// gating: one wave32 per token; E fixed at 8, top-2 (reference config)
__global__ void moe_gate(const float* __restrict__ x,
                         const float* __restrict__ gw,  // [C,8]
                         const float* __restrict__ gb,  // [8]
                         int2* __restrict__ lists,      // [8][N] (token, pBits)
                         int* __restrict__ counts,      // [8]
                         int C, int N) {
  const int wave = threadIdx.x >> 5;
  const int lane = threadIdx.x & 31;
  const int t = blockIdx.x * 8 + wave;
  if (t >= N) return;

  float acc[8];
#pragma unroll
  for (int e = 0; e < 8; ++e) acc[e] = 0.f;
  for (int k = lane; k < C; k += 32) {
    float xv = x[(long)t * C + k];
    const float4* gp = (const float4*)(gw + (long)k * 8);
    float4 g0 = gp[0], g1 = gp[1];
    acc[0] += xv * g0.x; acc[1] += xv * g0.y;
    acc[2] += xv * g0.z; acc[3] += xv * g0.w;
    acc[4] += xv * g1.x; acc[5] += xv * g1.y;
    acc[6] += xv * g1.z; acc[7] += xv * g1.w;
  }
#pragma unroll
  for (int off = 16; off > 0; off >>= 1) {
#pragma unroll
    for (int e = 0; e < 8; ++e) acc[e] += __shfl_xor(acc[e], off, 32);
  }
  if (lane == 0) {
    float lg[8];
#pragma unroll
    for (int e = 0; e < 8; ++e) lg[e] = acc[e] + gb[e];
    int i0 = 0; float v0 = lg[0];
#pragma unroll
    for (int e = 1; e < 8; ++e) if (lg[e] > v0) { v0 = lg[e]; i0 = e; }
    int i1 = -1; float v1 = -3.4e38f;
#pragma unroll
    for (int e = 0; e < 8; ++e)
      if (e != i0 && lg[e] > v1) { v1 = lg[e]; i1 = e; }
    float e1 = expf(v1 - v0);
    float p0 = 1.f / (1.f + e1);
    float p1 = e1 / (1.f + e1);
    int p = atomicAdd(&counts[i0], 1);
    lists[(long)i0 * N + p] = make_int2(t, __float_as_int(p0));
    p = atomicAdd(&counts[i1], 1);
    lists[(long)i1 * N + p] = make_int2(t, __float_as_int(p1));
  }
}

// ---------------------------------------------------------------------------
// GEMM1: H[r,:] = relu( X[token(r),:] @ W + bias ), bf16 out.
//   X: bf16 [Ntok][Kdim]; WT: bf16 [Nout][Kdim] (pre-transposed)
// ---------------------------------------------------------------------------
__global__ void __launch_bounds__(256)
moe_gemm1(const unsigned short* __restrict__ X, int lda,
          const unsigned short* __restrict__ WT,
          const float* __restrict__ bias,
          unsigned short* __restrict__ Hout, int ldh,
          const int* __restrict__ cntPtr, const int2* __restrict__ list,
          int Kdim, int Ntok) {
  const int cnt = cntPtr ? *cntPtr : Ntok;
  if ((int)blockIdx.x * TILE_M >= cnt) return;

  __shared__ __align__(16) unsigned short SM[2 * BUF_ELEMS];

  const int tid  = threadIdx.x;
  const int lane = tid & 31;
  const int wave = tid >> 5;
  const int rowBase = (wave >> 2) * 64;   // 0 / 64
  const int colBase = (wave & 3) * 64;    // 0..192
  const int lr = lane & 15;
  const int kh = (lane >> 4) << 3;        // A k-offset (elems) & C-row offset
  const int bh = (lane >> 4) << 4;        // unused; B k-offset is kh

  // ---- hoist all staging addresses out of the k-loop -----------------------
  const unsigned short* aSrc[2]; int aOff[2];
#pragma unroll
  for (int w = 0; w < 2; ++w) {                         // A: 128 rows x 32 k
    int c = tid + (w << 8);
    int row = c >> 2, kc = (c & 3) << 3;
    int gr = blockIdx.x * TILE_M + row;
    int srcRow = list ? ((gr < cnt) ? list[gr].x : 0) : gr;
    aSrc[w] = X + (size_t)srcRow * lda + kc;
    aOff[w] = row * LDSS + kc;
  }
  const unsigned short* bSrc[4]; int bOff[4];
#pragma unroll
  for (int w = 0; w < 4; ++w) {                         // B: 256 cols x 32 k
    int c = tid + (w << 8);
    int row = c >> 2, kc = (c & 3) << 3;
    bSrc[w] = WT + ((size_t)blockIdx.y * TILE_N + row) * Kdim + kc;
    bOff[w] = TILE_M * LDSS + row * LDSS + kc;
  }

  v8f acc[4][4];
#pragma unroll
  for (int i = 0; i < 4; ++i)
#pragma unroll
    for (int j = 0; j < 4; ++j) acc[i][j] = (v8f){0,0,0,0,0,0,0,0};

  const int nK = Kdim / TILE_K;
  auto stage = [&](int buf, int k0) {
    unsigned short* base = SM + buf * BUF_ELEMS;
#pragma unroll
    for (int w = 0; w < 2; ++w) ASYNC_CP(base + aOff[w], aSrc[w] + k0);
#pragma unroll
    for (int w = 0; w < 4; ++w) ASYNC_CP(base + bOff[w], bSrc[w] + k0);
  };

  stage(0, 0);
  for (int t = 0; t < nK; ++t) {
    WAIT_ASYNC0();
    __syncthreads();
    if (t + 1 < nK) stage((t + 1) & 1, (t + 1) * TILE_K);

    const unsigned short* Ab = SM + (t & 1) * BUF_ELEMS;
    const unsigned short* Bb = Ab + TILE_M * LDSS;
    FragU a[4], b[4];
#pragma unroll
    for (int i = 0; i < 4; ++i) {
      const unsigned short* p = Ab + (rowBase + i * 16 + lr) * LDSS + kh;
      a[i].q[0] = *(const uint4*)p;
      a[i].q[1] = *(const uint4*)(p + 16);
    }
#pragma unroll
    for (int j = 0; j < 4; ++j) {
      const unsigned short* p = Bb + (colBase + j * 16 + lr) * LDSS + kh;
      b[j].q[0] = *(const uint4*)p;
      b[j].q[1] = *(const uint4*)(p + 16);
    }
#pragma unroll
    for (int i = 0; i < 4; ++i)
#pragma unroll
      for (int j = 0; j < 4; ++j)
        acc[i][j] = __builtin_amdgcn_wmma_f32_16x16x32_bf16(
            false, a[i].v, false, b[j].v, (short)0, acc[i][j], false, false);
  }

  const int gcol0 = blockIdx.y * TILE_N + colBase;
  const bool full = (int)(blockIdx.x + 1) * TILE_M <= cnt;
  if (full) {
#pragma unroll
    for (int j = 0; j < 4; ++j) {
      int col = gcol0 + j * 16 + lr;
      float bb = bias[col];
#pragma unroll
      for (int i = 0; i < 4; ++i) {
        int rb = blockIdx.x * TILE_M + rowBase + i * 16 + kh;
#pragma unroll
        for (int g = 0; g < 8; ++g) {
          float v = acc[i][j][g] + bb;
          v = v > 0.f ? v : 0.f;
          Hout[(long)(rb + g) * ldh + col] = f2bf(v);
        }
      }
    }
  } else {
#pragma unroll
    for (int j = 0; j < 4; ++j) {
      int col = gcol0 + j * 16 + lr;
      float bb = bias[col];
#pragma unroll
      for (int i = 0; i < 4; ++i) {
        int rb = blockIdx.x * TILE_M + rowBase + i * 16 + kh;
#pragma unroll
        for (int g = 0; g < 8; ++g) {
          int r = rb + g;
          if (r < cnt) {
            float v = acc[i][j][g] + bb;
            v = v > 0.f ? v : 0.f;
            Hout[(long)r * ldh + col] = f2bf(v);
          }
        }
      }
    }
  }
}

// ---------------------------------------------------------------------------
// GEMM2: Out[token(r),:] (+)= p(r) * ( H[r,:] @ W + bias )
//   H: bf16 [Ntok][Kdim]; WT: bf16 [Nout][Kdim]; list==nullptr -> shared
// ---------------------------------------------------------------------------
__global__ void __launch_bounds__(256)
moe_gemm2(const unsigned short* __restrict__ Hin, int lda,
          const unsigned short* __restrict__ WT,
          const float* __restrict__ bias,
          float* __restrict__ Out, int ldo,
          const int* __restrict__ cntPtr, const int2* __restrict__ list,
          int Kdim, int Ntok, int accumulate) {
  const int cnt = cntPtr ? *cntPtr : Ntok;
  if ((int)blockIdx.x * TILE_M >= cnt) return;

  __shared__ __align__(16) unsigned short SM[2 * BUF_ELEMS];

  const int tid  = threadIdx.x;
  const int lane = tid & 31;
  const int wave = tid >> 5;
  const int rowBase = (wave >> 2) * 64;
  const int colBase = (wave & 3) * 64;
  const int lr = lane & 15;
  const int kh = (lane >> 4) << 3;
  const int bh = (lane >> 4) << 4;

  const unsigned short* aSrc[2]; int aOff[2];
#pragma unroll
  for (int w = 0; w < 2; ++w) {
    int c = tid + (w << 8);
    int row = c >> 2, kc = (c & 3) << 3;
    long gr = blockIdx.x * TILE_M + row;                // compact rows
    aSrc[w] = Hin + gr * lda + kc;
    aOff[w] = row * LDSS + kc;
  }
  const unsigned short* bSrc[4]; int bOff[4];
#pragma unroll
  for (int w = 0; w < 4; ++w) {
    int c = tid + (w << 8);
    int row = c >> 2, kc = (c & 3) << 3;
    bSrc[w] = WT + ((size_t)blockIdx.y * TILE_N + row) * Kdim + kc;
    bOff[w] = TILE_M * LDSS + row * LDSS + kc;
  }

  v8f acc[4][4];
#pragma unroll
  for (int i = 0; i < 4; ++i)
#pragma unroll
    for (int j = 0; j < 4; ++j) acc[i][j] = (v8f){0,0,0,0,0,0,0,0};

  const int nK = Kdim / TILE_K;
  auto stage = [&](int buf, int k0) {
    unsigned short* base = SM + buf * BUF_ELEMS;
#pragma unroll
    for (int w = 0; w < 2; ++w) ASYNC_CP(base + aOff[w], aSrc[w] + k0);
#pragma unroll
    for (int w = 0; w < 4; ++w) ASYNC_CP(base + bOff[w], bSrc[w] + k0);
  };

  stage(0, 0);
  for (int t = 0; t < nK; ++t) {
    WAIT_ASYNC0();
    __syncthreads();
    if (t + 1 < nK) stage((t + 1) & 1, (t + 1) * TILE_K);

    const unsigned short* Ab = SM + (t & 1) * BUF_ELEMS;
    const unsigned short* Bb = Ab + TILE_M * LDSS;
    FragU a[4], b[4];
#pragma unroll
    for (int i = 0; i < 4; ++i) {
      const unsigned short* p = Ab + (rowBase + i * 16 + lr) * LDSS + kh;
      a[i].q[0] = *(const uint4*)p;
      a[i].q[1] = *(const uint4*)(p + 16);
    }
#pragma unroll
    for (int j = 0; j < 4; ++j) {
      const unsigned short* p = Bb + (colBase + j * 16 + lr) * LDSS + kh;
      b[j].q[0] = *(const uint4*)p;
      b[j].q[1] = *(const uint4*)(p + 16);
    }
#pragma unroll
    for (int i = 0; i < 4; ++i)
#pragma unroll
      for (int j = 0; j < 4; ++j)
        acc[i][j] = __builtin_amdgcn_wmma_f32_16x16x32_bf16(
            false, a[i].v, false, b[j].v, (short)0, acc[i][j], false, false);
  }

  const int gcol0 = blockIdx.y * TILE_N + colBase;
  float bb[4];
#pragma unroll
  for (int j = 0; j < 4; ++j) bb[j] = bias[gcol0 + j * 16 + lr];

  const bool full = (int)(blockIdx.x + 1) * TILE_M <= cnt;
  if (full) {
#pragma unroll
    for (int i = 0; i < 4; ++i) {
      int rb = blockIdx.x * TILE_M + rowBase + i * 16 + kh;
#pragma unroll
      for (int g = 0; g < 8; ++g) {
        int r = rb + g;
        int tok; float wgt;
        if (list) { int2 le = list[r]; tok = le.x; wgt = __int_as_float(le.y); }
        else      { tok = r; wgt = 1.f; }
        long o = (long)tok * ldo + gcol0 + lr;
#pragma unroll
        for (int j = 0; j < 4; ++j) {
          float v = wgt * (acc[i][j][g] + bb[j]);
          if (accumulate) Out[o + j * 16] += v;
          else            Out[o + j * 16]  = v;
        }
      }
    }
  } else {
#pragma unroll
    for (int i = 0; i < 4; ++i) {
      int rb = blockIdx.x * TILE_M + rowBase + i * 16 + kh;
#pragma unroll
      for (int g = 0; g < 8; ++g) {
        int r = rb + g;
        if (r < cnt) {
          int tok; float wgt;
          if (list) { int2 le = list[r]; tok = le.x; wgt = __int_as_float(le.y); }
          else      { tok = r; wgt = 1.f; }
          long o = (long)tok * ldo + gcol0 + lr;
#pragma unroll
          for (int j = 0; j < 4; ++j) {
            float v = wgt * (acc[i][j][g] + bb[j]);
            if (accumulate) Out[o + j * 16] += v;
            else            Out[o + j * 16]  = v;
          }
        }
      }
    }
  }
}

// ---------------------------------------------------------------------------
// host launcher
// ---------------------------------------------------------------------------
extern "C" void kernel_launch(void* const* d_in, const int* in_sizes, int n_in,
                              void* d_out, int out_size, void* d_ws, size_t ws_size,
                              hipStream_t stream) {
  (void)n_in; (void)out_size; (void)ws_size;
  const float* x      = (const float*)d_in[0];
  const float* gate_w = (const float*)d_in[1];
  const float* gate_b = (const float*)d_in[2];
  const float* w1     = (const float*)d_in[3];
  const float* b1     = (const float*)d_in[4];
  const float* w2     = (const float*)d_in[5];
  const float* b2     = (const float*)d_in[6];
  const float* sw1    = (const float*)d_in[7];
  const float* sb1    = (const float*)d_in[8];
  const float* sw2    = (const float*)d_in[9];
  const float* sb2    = (const float*)d_in[10];
  float* out = (float*)d_out;

  const int E = in_sizes[2];                 // 8
  const int C = in_sizes[1] / E;             // 1024
  const int N = in_sizes[0] / C;             // 8192
  const int H = in_sizes[4] / E;             // 4096

  // ws: counts | lists | x_bf16 | h_bf16 | w1T | w2T | sw1T | sw2T
  auto al = [](size_t v) { return (v + 255) & ~(size_t)255; };
  char* ws = (char*)d_ws;
  size_t off = 0;
  int* counts = (int*)(ws + off);                    off = al(off + (size_t)E * 4);
  int2* lists = (int2*)(ws + off);                   off = al(off + (size_t)E * N * 8);
  unsigned short* xbf  = (unsigned short*)(ws + off); off = al(off + (size_t)N * C * 2);
  unsigned short* hws  = (unsigned short*)(ws + off); off = al(off + (size_t)N * H * 2);
  unsigned short* w1T  = (unsigned short*)(ws + off); off = al(off + (size_t)E * C * H * 2);
  unsigned short* w2T  = (unsigned short*)(ws + off); off = al(off + (size_t)E * H * C * 2);
  unsigned short* sw1T = (unsigned short*)(ws + off); off = al(off + (size_t)C * H * 2);
  unsigned short* sw2T = (unsigned short*)(ws + off); off = al(off + (size_t)H * C * 2);

  moe_zero_counts<<<1, 64, 0, stream>>>(counts, E);
  long n8 = (long)N * C / 8;
  moe_f32_to_bf16<<<dim3((unsigned)((n8 + 255) / 256)), dim3(256), 0, stream>>>(x, xbf, n8);
  moe_gate<<<dim3((N + 7) / 8), dim3(256), 0, stream>>>(x, gate_w, gate_b, lists, counts, C, N);

  dim3 tb(32, 8);
  // sw1 [C][H] -> sw1T [H][C]; sw2 [H][C] -> sw2T [C][H]
  moe_transpose_bf16<<<dim3(H / 32, C / 32, 1), tb, 0, stream>>>(sw1, sw1T, C, H);
  moe_transpose_bf16<<<dim3(C / 32, H / 32, 1), tb, 0, stream>>>(sw2, sw2T, H, C);
  moe_transpose_bf16<<<dim3(H / 32, C / 32, E), tb, 0, stream>>>(w1, w1T, C, H);
  moe_transpose_bf16<<<dim3(C / 32, H / 32, E), tb, 0, stream>>>(w2, w2T, H, C);

  dim3 blk(256);
  dim3 g1(N / TILE_M, H / TILE_N);   // 64 x 16
  dim3 g2(N / TILE_M, C / TILE_N);   // 64 x 4

  // shared expert first (plain store initializes d_out), then experts 0..E-1
  moe_gemm1<<<g1, blk, 0, stream>>>(xbf, C, sw1T, sb1, hws, H,
                                    nullptr, nullptr, C, N);
  moe_gemm2<<<g2, blk, 0, stream>>>(hws, H, sw2T, sb2, out, C,
                                    nullptr, nullptr, H, N, 0);
  for (int e = 0; e < E; ++e) {
    moe_gemm1<<<g1, blk, 0, stream>>>(xbf, C, w1T + (size_t)e * C * H,
                                      b1 + (size_t)e * H, hws, H,
                                      counts + e, lists + (size_t)e * N, C, N);
    moe_gemm2<<<g2, blk, 0, stream>>>(hws, H, w2T + (size_t)e * H * C,
                                      b2 + (size_t)e * C, out, C,
                                      counts + e, lists + (size_t)e * N, H, N, 1);
  }
}
